// LSTM_43267500540292
// MI455X (gfx1250) — matrix operand order repaired and checked by reading, not compiled, not run on hardware
//
#include <hip/hip_runtime.h>

// ---------------------------------------------------------------------------
// LSTM on MI455X (gfx1250): bf16 WMMA recurrent GEMM, fused gate epilogue.
// B=64, T=512, I=512, H=1024.
// Per step: gates = x_t @ Wx^T + h @ Wh^T + b (4 gates, fused K = I+H = 1536),
// then lane-local LSTM pointwise update. One kernel launch per timestep.
// Depth-2 software pipeline enforced with __builtin_amdgcn_sched_barrier.
// ---------------------------------------------------------------------------

typedef __attribute__((ext_vector_type(16))) __bf16 v16bf;
typedef __attribute__((ext_vector_type(8)))  __bf16 v8bf;
typedef __attribute__((ext_vector_type(8)))  float  v8f;

static constexpr int Bn = 64;    // batch
static constexpr int Tn = 512;   // time
static constexpr int In = 512;   // input dim
static constexpr int Hn = 1024;  // hidden dim
static constexpr int NCHUNK = (In + Hn) / 32;   // 48 K-chunks of 32

#if defined(__has_builtin)
#if __has_builtin(__builtin_amdgcn_sched_barrier)
#define SCHED_FENCE() __builtin_amdgcn_sched_barrier(0)
#endif
#endif
#ifndef SCHED_FENCE
#define SCHED_FENCE() ((void)0)
#endif

// ---- fp32 -> bf16 (round-to-nearest-even) ---------------------------------
__device__ __forceinline__ unsigned short f2bf(float f) {
    unsigned int u = __float_as_uint(f);
    u += 0x7FFFu + ((u >> 16) & 1u);
    return (unsigned short)(u >> 16);
}

__global__ void cvt_f32_to_bf16(const float* __restrict__ in,
                                unsigned short* __restrict__ out, int n) {
    int i = blockIdx.x * blockDim.x + threadIdx.x;
    int stride = gridDim.x * blockDim.x;
    for (; i < n; i += stride) out[i] = f2bf(in[i]);
}

__global__ void init_state(unsigned short* __restrict__ hbuf,
                           float* __restrict__ cbuf, int n) {
    int i = blockIdx.x * blockDim.x + threadIdx.x;
    if (i < n) { hbuf[i] = 0; cbuf[i] = 0.0f; }
}

__device__ __forceinline__ float sigmoidf_(float x) {
    return 1.0f / (1.0f + __expf(-x));
}

__device__ __forceinline__ v16bf join8(v8bf lo, v8bf hi) {
    return __builtin_shufflevector(lo, hi, 0, 1, 2, 3, 4, 5, 6, 7,
                                   8, 9, 10, 11, 12, 13, 14, 15);
}

struct FragSet {
    v16bf a, b0, b1, b2, b3;
};

// ---------------------------------------------------------------------------
// One timestep.
// Grid: Hn/16 = 64 blocks, 128 threads (4 wave32) -> 256 waves total.
// Block owns one 16-column slice [16*blk, 16*blk+16) for ALL four gates.
// Wave w -> mTile = w (16 batch rows).  acc[g] = 4 WMMA C/D tiles (32 VGPRs).
// ---------------------------------------------------------------------------
__global__ __launch_bounds__(128)
void lstm_step(const unsigned short* __restrict__ xbf,   // (B,T,I) bf16
               const unsigned short* __restrict__ Wx,    // 4 x (H,I) bf16
               const unsigned short* __restrict__ Wh,    // 4 x (H,H) bf16
               const float* __restrict__ b_i, const float* __restrict__ b_f,
               const float* __restrict__ b_c, const float* __restrict__ b_o,
               unsigned short* __restrict__ hbuf,        // (B,H) bf16 state
               float* __restrict__ cbuf,                 // (B,H) f32 state
               float* __restrict__ out,                  // outputs ++ h_n ++ c_n
               int t, int last) {
    const int lane  = threadIdx.x & 31;
    const int wave  = threadIdx.x >> 5;   // 0..3 -> mTile
    const int moff  = wave * 16;
    const int nBase = blockIdx.x * 16;

    const int lm     = lane & 15;
    const int hiHalf = (lane >> 4) & 1;   // lanes 16..31 -> second K half
    const int aShift = hiHalf * 16;       // A-fragment K sub-offset (elements)
    const int bShift = hiHalf * 16;       // B-fragment K sub-offset (elements)

    // Precomputed row bases (element offsets)
    const size_t xrow = ((size_t)(moff + lm) * Tn + t) * In;  // x[b][t][*]
    const size_t hrow = (size_t)(moff + lm) * Hn;             // h[b][*]
    const size_t wrowI = (size_t)(nBase + lm) * In;           // Wx[g][n][*]
    const size_t wrowH = (size_t)(nBase + lm) * Hn;           // Wh[g][n][*]

    // Chunk c in [0,48): c<16 -> x/Wx (k0 = 32c), else h/Wh (k0 = 32(c-16)).
    auto loadA = [&](int c) -> v16bf {
        const unsigned short* ap =
            (c < 16) ? (xbf + xrow + c * 32 + aShift)
                     : (hbuf + hrow + (c - 16) * 32 + aShift);
        v8bf lo = *(const v8bf*)(ap);
        v8bf hi = *(const v8bf*)(ap + 8);
        return join8(lo, hi);
    };
    auto loadB = [&](int c, int g) -> v16bf {
        const unsigned short* bp =
            (c < 16) ? (Wx + (size_t)g * Hn * In + wrowI + c * 32 + bShift)
                     : (Wh + (size_t)g * Hn * Hn + wrowH + (c - 16) * 32 + bShift);
        return *(const v16bf*)bp;
    };
    auto loadSet = [&](int c) -> FragSet {
        FragSet s;
        s.a = loadA(c);
        s.b0 = loadB(c, 0);
        s.b1 = loadB(c, 1);
        s.b2 = loadB(c, 2);
        s.b3 = loadB(c, 3);
        return s;
    };

    v8f acc[4];
    const v8f vzero = {0.f, 0.f, 0.f, 0.f, 0.f, 0.f, 0.f, 0.f};
#pragma unroll
    for (int g = 0; g < 4; ++g) acc[g] = vzero;

    // ---- depth-2 modulo-scheduled K loop ----------------------------------
    // Iteration c: issue loads for chunk c+2, fence, WMMA on chunk c
    // (fragments loaded two iterations earlier -> two chunks of latency cover).
    FragSet s0 = loadSet(0);
    FragSet s1 = loadSet(1);

#pragma unroll
    for (int c = 0; c < NCHUNK; ++c) {
        FragSet s2 = s0;                   // dead copy on the tail iterations
        if (c + 2 < NCHUNK) s2 = loadSet(c + 2);
        SCHED_FENCE();                     // keep prefetch ahead of the WMMAs
        acc[0] = __builtin_amdgcn_wmma_f32_16x16x32_bf16(
            false, s0.a, false, s0.b0, (short)0, acc[0], false, false);
        acc[1] = __builtin_amdgcn_wmma_f32_16x16x32_bf16(
            false, s0.a, false, s0.b1, (short)0, acc[1], false, false);
        acc[2] = __builtin_amdgcn_wmma_f32_16x16x32_bf16(
            false, s0.a, false, s0.b2, (short)0, acc[2], false, false);
        acc[3] = __builtin_amdgcn_wmma_f32_16x16x32_bf16(
            false, s0.a, false, s0.b3, (short)0, acc[3], false, false);
        s0 = s1;
        s1 = s2;
    }

    // ---- Epilogue: LSTM pointwise update (lane-local) ---------------------
    // C/D layout: VGPR r holds (M = moff + r + 8*hiHalf, N = nBase + lm)
    const int n = nBase + lm;
    const float bi = b_i[n], bf = b_f[n], bc = b_c[n], bo = b_o[n];
#pragma unroll
    for (int r = 0; r < 8; ++r) {
        const int b = moff + r + 8 * hiHalf;
        const float it = sigmoidf_(acc[0][r] + bi);
        const float ft = sigmoidf_(acc[1][r] + bf);
        const float ct = tanhf(acc[2][r] + bc);
        const float ot = sigmoidf_(acc[3][r] + bo);
        const size_t sidx = (size_t)b * Hn + n;
        const float cold = cbuf[sidx];
        const float cnew = ft * cold + it * ct;
        const float hnew = ot * tanhf(cnew);
        cbuf[sidx] = cnew;
        hbuf[sidx] = f2bf(hnew);
        out[(size_t)b * Tn * Hn + (size_t)t * Hn + n] = hnew;
        if (last) {
            const size_t tail = (size_t)Bn * Tn * Hn;
            out[tail + sidx] = hnew;                     // h_n
            out[tail + (size_t)Bn * Hn + sidx] = cnew;   // c_n
        }
    }

    // Warm next timestep's x rows into cache for the next launch
    // (emits global_prefetch_b8; address valid whenever t+1 < Tn).
    if (t + 1 < Tn) {
        const unsigned short* px =
            xbf + ((size_t)(moff + lm) * Tn + (t + 1)) * In + hiHalf * 256;
        __builtin_prefetch(px, 0, 1);
    }
}

// ---------------------------------------------------------------------------
// Host launcher.  Workspace layout (bytes), total ~44.4 MB:
//   [0)            xbf   : B*T*I bf16  = 33,554,432
//   [33554432)     Wx    : 4*H*I bf16  =  4,194,304
//   [37748736)     Wh    : 4*H*H bf16  =  8,388,608
//   [46137344)     hbuf  : B*H bf16    =    131,072
//   [46268416)     cbuf  : B*H f32     =    262,144
// ---------------------------------------------------------------------------
extern "C" void kernel_launch(void* const* d_in, const int* in_sizes, int n_in,
                              void* d_out, int out_size, void* d_ws, size_t ws_size,
                              hipStream_t stream) {
    (void)in_sizes; (void)n_in; (void)out_size; (void)ws_size;

    const float* x = (const float*)d_in[0];
    // gate order: 0=i, 1=f, 2=c, 3=o
    const float* Wxf32[4] = {(const float*)d_in[1], (const float*)d_in[4],
                             (const float*)d_in[7], (const float*)d_in[10]};
    const float* Whf32[4] = {(const float*)d_in[2], (const float*)d_in[5],
                             (const float*)d_in[8], (const float*)d_in[11]};
    const float* bias[4]  = {(const float*)d_in[3], (const float*)d_in[6],
                             (const float*)d_in[9], (const float*)d_in[12]};

    char* ws = (char*)d_ws;
    unsigned short* xbf  = (unsigned short*)(ws);
    unsigned short* Wx   = (unsigned short*)(ws + 33554432u);
    unsigned short* Wh   = (unsigned short*)(ws + 37748736u);
    unsigned short* hbuf = (unsigned short*)(ws + 46137344u);
    float*          cbuf = (float*)(ws + 46268416u);
    float*          out  = (float*)d_out;

    // fp32 -> bf16 conversions
    cvt_f32_to_bf16<<<4096, 256, 0, stream>>>(x, xbf, Bn * Tn * In);
    for (int g = 0; g < 4; ++g)
        cvt_f32_to_bf16<<<512, 256, 0, stream>>>(Wxf32[g], Wx + (size_t)g * Hn * In,
                                                 Hn * In);
    for (int g = 0; g < 4; ++g)
        cvt_f32_to_bf16<<<1024, 256, 0, stream>>>(Whf32[g], Wh + (size_t)g * Hn * Hn,
                                                  Hn * Hn);

    // zero initial h, c
    init_state<<<(Bn * Hn + 255) / 256, 256, 0, stream>>>(hbuf, cbuf, Bn * Hn);

    // sequential recurrence: one fused kernel per timestep
    for (int t = 0; t < Tn; ++t) {
        lstm_step<<<Hn / 16, 128, 0, stream>>>(
            xbf, Wx, Wh, bias[0], bias[1], bias[2], bias[3],
            hbuf, cbuf, out, t, (t == Tn - 1) ? 1 : 0);
    }
}
